// RNN_Classifier_47983374631465
// MI455X (gfx1250) — compile-verified, matrix operand order checked by reading, not run on hardware
//
#include <hip/hip_runtime.h>

typedef __attribute__((ext_vector_type(16))) _Float16 v16h;
typedef __attribute__((ext_vector_type(8)))  float    v8f;
typedef unsigned int u32x4 __attribute__((ext_vector_type(4)));
typedef int          i32x8 __attribute__((ext_vector_type(8)));
typedef int          i32x4 __attribute__((ext_vector_type(4)));

// ---------------------------------------------------------------------------
// WMMA fp16 fragment loader (A and B share the same per-lane layout on CDNA5):
// lane L holds row (base + L%16); halves K+{0..7,16..23} for L<16,
// K+{8..15,24..31} for L>=16  ->  two 16-byte contiguous loads.
// ---------------------------------------------------------------------------
__device__ __forceinline__ v16h ldfrag16(const _Float16* base, int rowBase, int stride, int k0)
{
    const int lane = threadIdx.x & 31;
    const int r  = rowBase + (lane & 15);
    const int kk = k0 + ((lane & 16) ? 8 : 0);
    const _Float16* p = base + (size_t)r * stride + kk;
    union { v16h h; float4 f[2]; } u;
    u.f[0] = *(const float4*)(p);
    u.f[1] = *(const float4*)(p + 16);
    return u.h;
}

__device__ __forceinline__ float sigmoidf_(float x) { return 1.0f / (1.0f + __expf(-x)); }

// ---------------------------------------------------------------------------
// TDM: async DMA of one contiguous slice (global -> LDS), 8-byte elements.
// D# packed per CDNA5 ISA 8.3/8.4; groups 2/3 zero (tensor <= 2D).
// 6-arg builtin form (clang-23 / therock-10.0 toolchain).
// ---------------------------------------------------------------------------
__device__ __forceinline__ void tdm_load_slice(const _Float16* gsrc, unsigned ldsByteOff, int bytes)
{
    const unsigned long long ga = (unsigned long long)(const void*)gsrc;
    const int n8 = bytes >> 3;                       // 8-byte elements
    u32x4 g0;
    g0[0] = 1u;                                      // count=1 (valid user D#)
    g0[1] = ldsByteOff;                              // lds_addr (bytes)
    g0[2] = (unsigned)(ga & 0xffffffffull);          // global_addr[31:0]  -> bits[95:64]
    g0[3] = (unsigned)((ga >> 32) & 0x1ffffffull)    // global_addr[56:32] -> bits[120:96]
          | (2u << 30);                              // type=2 ("image")
    i32x8 g1;
    g1[0] = 3 << 16;                                 // wg_mask=0, data_size=3 (8B)
    g1[1] = (n8 & 0xffff) << 16;                     // tensor_dim0[15:0] @ bits[63:48]
    g1[2] = ((n8 >> 16) & 0xffff) | (1 << 16);       // tensor_dim0[31:16]; tensor_dim1=1
    g1[3] = (n8 & 0xffff) << 16;                     // tile_dim0 = n8 @ bits[127:112]
    g1[4] = 1;                                       // tile_dim1 = 1, tile_dim2 = 0
    g1[5] = n8;                                      // tensor_dim0_stride (single row)
    g1[6] = 0;
    g1[7] = 0;
    const i32x4 z4 = {0, 0, 0, 0};
    const i32x8 z8 = {0, 0, 0, 0, 0, 0, 0, 0};
    __builtin_amdgcn_tensor_load_to_lds(g0, g1, z4, z4, z8, 0);
}

// ---------------------------------------------------------------------------
// G = X * Wih^T + (bih+bhh), both directions.  X:[M][Kpad] f16 row-major,
// W:[2][N][Kpad] f16 (row n of W == column n of B operand), G:[2][M][N] f16.
// One 16x16 output tile per wave, 8 waves per block.
// ---------------------------------------------------------------------------
__global__ __launch_bounds__(256) void xw_gemm_kernel(
        const _Float16* __restrict__ X,
        const _Float16* __restrict__ W,
        const float*    __restrict__ bias,   // [2][N]
        _Float16*       __restrict__ G,
        int M, int N, int Kpad)
{
    const int wave = threadIdx.x >> 5;
    const int lane = threadIdx.x & 31;
    const int mt   = blockIdx.x * 8 + wave;
    const int nt   = blockIdx.y % (N >> 4);
    const int dir  = blockIdx.y / (N >> 4);

    const _Float16* Wd = W + (size_t)dir * N * Kpad;
    v8f acc = {};
    for (int k0 = 0; k0 < Kpad; k0 += 32) {
        v16h a = ldfrag16(X,  mt * 16, Kpad, k0);
        v16h b = ldfrag16(Wd, nt * 16, Kpad, k0);
        acc = __builtin_amdgcn_wmma_f32_16x16x32_f16(false, a, false, b, (short)0, acc, false, false);
    }
    const int n  = nt * 16 + (lane & 15);
    const int mb = mt * 16 + ((lane & 16) ? 8 : 0);
    const float bv = bias[dir * N + n];
    _Float16* gout = G + (size_t)dir * M * N;
#pragma unroll
    for (int v = 0; v < 8; ++v)
        gout[(size_t)(mb + v) * N + n] = (_Float16)(acc[v] + bv);
}

// ---------------------------------------------------------------------------
// Persistent recurrent scan: one block per direction.  Whh + h + double-
// buffered G[t] slice live in LDS (<=272KB for H=128; WGP has 320KB).
// Each wave owns (batch-tile, h-column-tile) groups and computes all FOUR
// gate tiles for its group (A fragment reused across 4 WMMAs), so the LSTM
// pointwise update runs on registers; c stays in VGPRs for the whole scan.
// TDM prefetches the next step's G slice while this step computes.
// ---------------------------------------------------------------------------
template<int H>
__global__ __launch_bounds__(512) void lstm_scan_kernel(
        const _Float16* __restrict__ Whh,   // [2][4H][Kpad] f16
        const _Float16* __restrict__ G,     // [2][S*B][4H]  f16
        float*          __restrict__ Y,     // [S*B][2H]     f32
        int S)
{
    constexpr int Bb     = 64;
    constexpr int Kpad   = (H < 32) ? 32 : H;
    constexpr int FourH  = 4 * H;
    constexpr int MT     = Bb / 16;              // 4 batch tiles
    constexpr int HT     = H / 16;               // h-column tiles
    constexpr int GROUPS = MT * HT;              // (mt,ht) groups
    constexpr int GPW    = (GROUPS + 15) / 16;   // groups per wave (16 waves)
    constexpr int sliceElems = Bb * FourH;
    constexpr int sliceBytes = sliceElems * 2;

    extern __shared__ char lds[];
    _Float16* Wh   = (_Float16*)lds;             // [4H][Kpad]
    _Float16* hbuf = Wh + FourH * Kpad;          // [Bb][Kpad]
    _Float16* gbuf = hbuf + Bb * Kpad;           // [2][Bb*4H]  (TDM target)
    const unsigned gbufOff = (unsigned)((FourH * Kpad + Bb * Kpad) * sizeof(_Float16));

    const int dir  = blockIdx.x;
    const int tid  = threadIdx.x;
    const int wave = tid >> 5;
    const int lane = tid & 31;

    const _Float16* Wsrc = Whh + (size_t)dir * FourH * Kpad;
    for (int i = tid; i < FourH * Kpad; i += blockDim.x) Wh[i] = Wsrc[i];
    for (int i = tid; i < Bb * Kpad;    i += blockDim.x) hbuf[i] = (_Float16)0.f;
    __syncthreads();

    const _Float16* Gd = G + (size_t)dir * S * sliceElems;
    if (wave == 0) {                                    // preload first step
        const int t0 = dir ? (S - 1) : 0;
        tdm_load_slice(Gd + (size_t)t0 * sliceElems, gbufOff, sliceBytes);
    }

    float creg[GPW][8];
#pragma unroll
    for (int j = 0; j < GPW; ++j)
#pragma unroll
        for (int v = 0; v < 8; ++v) creg[j][v] = 0.f;

    const int n  = lane & 15;
    const int mb = (lane & 16) ? 8 : 0;

    for (int s = 0; s < S; ++s) {
        const int t = dir ? (S - 1 - s) : s;

        if (wave == 0) __builtin_amdgcn_s_wait_tensorcnt(0);
        __syncthreads();   // gbuf[s&1] landed; prev step's hbuf writes visible

        if (wave == 0 && s + 1 < S) {                   // async-prefetch next step
            const int t1 = dir ? (t - 1) : (t + 1);
            tdm_load_slice(Gd + (size_t)t1 * sliceElems,
                           gbufOff + (unsigned)(((s + 1) & 1) * sliceBytes), sliceBytes);
        }
        const _Float16* gc = gbuf + (s & 1) * sliceElems;

        float hreg[GPW][8];
#pragma unroll
        for (int j = 0; j < GPW; ++j) {
            const int grp = wave * GPW + j;
            if (grp < GROUPS) {
                const int mt = grp % MT, ht = grp / MT;
                v8f acc0 = {}, acc1 = {}, acc2 = {}, acc3 = {};
#pragma unroll
                for (int k0 = 0; k0 < Kpad; k0 += 32) {
                    v16h a  = ldfrag16(hbuf, mt * 16, Kpad, k0);
                    v16h b0 = ldfrag16(Wh, 0 * H + ht * 16, Kpad, k0);
                    v16h b1 = ldfrag16(Wh, 1 * H + ht * 16, Kpad, k0);
                    v16h b2 = ldfrag16(Wh, 2 * H + ht * 16, Kpad, k0);
                    v16h b3 = ldfrag16(Wh, 3 * H + ht * 16, Kpad, k0);
                    acc0 = __builtin_amdgcn_wmma_f32_16x16x32_f16(false, a, false, b0, (short)0, acc0, false, false);
                    acc1 = __builtin_amdgcn_wmma_f32_16x16x32_f16(false, a, false, b1, (short)0, acc1, false, false);
                    acc2 = __builtin_amdgcn_wmma_f32_16x16x32_f16(false, a, false, b2, (short)0, acc2, false, false);
                    acc3 = __builtin_amdgcn_wmma_f32_16x16x32_f16(false, a, false, b3, (short)0, acc3, false, false);
                }
                const int hh = ht * 16 + n;
#pragma unroll
                for (int v = 0; v < 8; ++v) {
                    const int bi = mt * 16 + mb + v;
                    const float gi = sigmoidf_(acc0[v] + (float)gc[bi * FourH + hh]);
                    const float gf = sigmoidf_(acc1[v] + (float)gc[bi * FourH + H + hh]);
                    const float gg = tanhf    (acc2[v] + (float)gc[bi * FourH + 2 * H + hh]);
                    const float go = sigmoidf_(acc3[v] + (float)gc[bi * FourH + 3 * H + hh]);
                    const float c  = gf * creg[j][v] + gi * gg;
                    creg[j][v] = c;
                    hreg[j][v] = go * tanhf(c);
                }
            }
        }
        __syncthreads();   // all waves finished reading old hbuf

#pragma unroll
        for (int j = 0; j < GPW; ++j) {
            const int grp = wave * GPW + j;
            if (grp < GROUPS) {
                const int mt = grp % MT, ht = grp / MT;
                const int hh = ht * 16 + n;
#pragma unroll
                for (int v = 0; v < 8; ++v) {
                    const int bi = mt * 16 + mb + v;
                    hbuf[bi * Kpad + hh] = (_Float16)hreg[j][v];
                    Y[((size_t)t * Bb + bi) * (2 * H) + dir * H + hh] = hreg[j][v];
                }
            }
        }
        // next iteration's top barrier orders these writes before the reads
    }
}

// ---------------------------------------------------------------------------
// BatchNorm (training batch stats) over M=S*B rows per channel; emits fp16
// activations for the next layer's WMMA GEMM.
// ---------------------------------------------------------------------------
__global__ __launch_bounds__(256) void bn_kernel(
        const float* __restrict__ Y, const float* __restrict__ gamma,
        const float* __restrict__ beta, _Float16* __restrict__ Xh, int M, int C)
{
    const int c = blockIdx.x, tid = threadIdx.x;
    __shared__ float s1[256], s2[256];
    float a = 0.f, b = 0.f;
    for (int i = tid; i < M; i += 256) { float v = Y[(size_t)i * C + c]; a += v; b += v * v; }
    s1[tid] = a; s2[tid] = b; __syncthreads();
    for (int st = 128; st > 0; st >>= 1) {
        if (tid < st) { s1[tid] += s1[tid + st]; s2[tid] += s2[tid + st]; }
        __syncthreads();
    }
    const float mean = s1[0] / (float)M;
    const float var  = s2[0] / (float)M - mean * mean;
    const float sc = gamma[c] * rsqrtf(var + 1e-5f);
    const float sh = beta[c] - mean * sc;
    for (int i = tid; i < M; i += 256)
        Xh[(size_t)i * C + c] = (_Float16)(Y[(size_t)i * C + c] * sc + sh);
}

// f32 [R][C] -> f16 [R][Cpad] (zero pad)
__global__ void convert_pad_kernel(const float* __restrict__ src, _Float16* __restrict__ dst,
                                   int R, int C, int Cpad)
{
    const long n = (long)R * Cpad;
    for (long i = blockIdx.x * (long)blockDim.x + threadIdx.x; i < n; i += (long)gridDim.x * blockDim.x) {
        const int r = (int)(i / Cpad), c = (int)(i % Cpad);
        dst[i] = (c < C) ? (_Float16)src[(long)r * C + c] : (_Float16)0.f;
    }
}

__global__ void bias_sum_kernel(const float* __restrict__ a, const float* __restrict__ b,
                                float* __restrict__ o, int n)
{
    for (int i = blockIdx.x * blockDim.x + threadIdx.x; i < n; i += gridDim.x * blockDim.x)
        o[i] = a[i] + b[i];
}

// x f32 [B][S][I] -> Xh f16 [S*B][Kpad]
__global__ void input_pack_kernel(const float* __restrict__ x, _Float16* __restrict__ Xh,
                                  int Bb, int S, int I, int Kpad)
{
    const long n = (long)S * Bb * Kpad;
    for (long i = blockIdx.x * (long)blockDim.x + threadIdx.x; i < n; i += (long)gridDim.x * blockDim.x) {
        const int c = (int)(i % Kpad);
        const long row = i / Kpad;
        const int b = (int)(row % Bb), s = (int)(row / Bb);
        Xh[i] = (c < I) ? (_Float16)x[((long)b * S + s) * I + c] : (_Float16)0.f;
    }
}

// out[b][s][o] = Xh[s*B+b][:] . fc_w[o][:] + fc_b[o]
__global__ void fc_kernel(const _Float16* __restrict__ Xh, const float* __restrict__ fw,
                          const float* __restrict__ fb, float* __restrict__ out, int S, int Bb)
{
    const long tot = (long)S * Bb * 11;
    for (long i = blockIdx.x * (long)blockDim.x + threadIdx.x; i < tot; i += (long)gridDim.x * blockDim.x) {
        const int o = (int)(i % 11);
        const long r = i / 11;
        const int b = (int)(r % Bb), s = (int)(r / Bb);
        const _Float16* xp = Xh + ((long)s * Bb + b) * 256;
        const float* wp = fw + o * 256;
        float acc = fb[o];
        for (int k = 0; k < 256; ++k) acc += (float)xp[k] * wp[k];
        out[((long)b * S + s) * 11 + o] = acc;
    }
}

// ---------------------------------------------------------------------------
extern "C" void kernel_launch(void* const* d_in, const int* in_sizes, int n_in,
                              void* d_out, int out_size, void* d_ws, size_t ws_size,
                              hipStream_t stream)
{
    (void)in_sizes; (void)n_in; (void)out_size; (void)ws_size;
    const int S = 2048, Bb = 64;
    const size_t Mrows = (size_t)S * Bb;

    char* ws = (char*)d_ws;
    size_t off = 0;
    auto carve = [&](size_t bytes) -> char* {
        char* p = ws + off;
        off = (off + bytes + 255) & ~(size_t)255;
        return p;
    };

    _Float16* Xh    = (_Float16*)carve(Mrows * 256 * sizeof(_Float16)); // activations (f16)
    float*    Yb    = (float*)   carve(Mrows * 256 * sizeof(float));    // scan output (f32)
    _Float16* G     = (_Float16*)carve(2 * Mrows * 512 * sizeof(_Float16));
    _Float16* WihH  = (_Float16*)carve(2 * 512 * 128 * sizeof(_Float16));
    _Float16* WhhH  = (_Float16*)carve(2 * 512 * 128 * sizeof(_Float16));
    float*    biasS = (float*)   carve(2 * 512 * sizeof(float));

    // layer-0 input: [B,S,7] f32 -> [S*B,32] f16 (zero-padded K)
    input_pack_kernel<<<dim3(4096), dim3(256), 0, stream>>>((const float*)d_in[0], Xh, Bb, S, 7, 32);

    const int Hs[4] = {16, 32, 64, 128};
    const int Is[4] = {7, 32, 64, 128};

    for (int li = 0; li < 4; ++li) {
        const int I = Is[li], H = Hs[li];
        const int N = 4 * H, C = 2 * H;
        const int KpadI = (I < 32) ? 32 : I;
        const int KpadH = (H < 32) ? 32 : H;
        const int base = 1 + 6 * li;
        const float* wih = (const float*)d_in[base + 0];
        const float* whh = (const float*)d_in[base + 1];
        const float* bih = (const float*)d_in[base + 2];
        const float* bhh = (const float*)d_in[base + 3];
        const float* gam = (const float*)d_in[base + 4];
        const float* bet = (const float*)d_in[base + 5];

        convert_pad_kernel<<<dim3(256), dim3(256), 0, stream>>>(wih, WihH, 2 * N, I, KpadI);
        convert_pad_kernel<<<dim3(256), dim3(256), 0, stream>>>(whh, WhhH, 2 * N, H, KpadH);
        bias_sum_kernel<<<dim3(8), dim3(256), 0, stream>>>(bih, bhh, biasS, 2 * N);

        xw_gemm_kernel<<<dim3((int)(Mrows / 16 / 8), 2 * (N / 16)), dim3(256), 0, stream>>>(
            Xh, WihH, biasS, G, (int)Mrows, N, KpadI);

        const size_t shmem = (size_t)N * KpadH * 2   // Whh
                           + (size_t)Bb * KpadH * 2  // h
                           + (size_t)2 * Bb * N * 2; // G double buffer (f16)
        switch (H) {
        case 16:
            (void)hipFuncSetAttribute(reinterpret_cast<const void*>(lstm_scan_kernel<16>),
                                      hipFuncAttributeMaxDynamicSharedMemorySize, (int)shmem);
            lstm_scan_kernel<16><<<dim3(2), dim3(512), shmem, stream>>>(WhhH, G, Yb, S);
            break;
        case 32:
            (void)hipFuncSetAttribute(reinterpret_cast<const void*>(lstm_scan_kernel<32>),
                                      hipFuncAttributeMaxDynamicSharedMemorySize, (int)shmem);
            lstm_scan_kernel<32><<<dim3(2), dim3(512), shmem, stream>>>(WhhH, G, Yb, S);
            break;
        case 64:
            (void)hipFuncSetAttribute(reinterpret_cast<const void*>(lstm_scan_kernel<64>),
                                      hipFuncAttributeMaxDynamicSharedMemorySize, (int)shmem);
            lstm_scan_kernel<64><<<dim3(2), dim3(512), shmem, stream>>>(WhhH, G, Yb, S);
            break;
        default:
            (void)hipFuncSetAttribute(reinterpret_cast<const void*>(lstm_scan_kernel<128>),
                                      hipFuncAttributeMaxDynamicSharedMemorySize, (int)shmem);
            lstm_scan_kernel<128><<<dim3(2), dim3(512), shmem, stream>>>(WhhH, G, Yb, S);
            break;
        }

        bn_kernel<<<dim3(C), dim3(256), 0, stream>>>(Yb, gam, bet, Xh, (int)Mrows, C);
    }

    fc_kernel<<<dim3(4096), dim3(256), 0, stream>>>(
        Xh, (const float*)d_in[25], (const float*)d_in[26], (float*)d_out, S, Bb);
}
